// TemporalSelfAttention_18545668784594
// MI455X (gfx1250) — matrix operand order repaired and checked by reading, not compiled, not run on hardware
//
#include <hip/hip_runtime.h>
#include <hip/hip_bf16.h>

#define EMB 256
#define NH 8
#define HD 32
#define NP 4
#define NSS 2
#define BSTR 264  // LDS row stride (elements): 528B = 4-bank skew -> conflict-free b128 reads

typedef __attribute__((ext_vector_type(16))) __bf16 v16bf;
typedef __attribute__((ext_vector_type(8)))  float  v8f;
typedef __attribute__((ext_vector_type(4)))  int    v4i;

#if defined(__has_builtin)
#  if __has_builtin(__builtin_amdgcn_global_load_async_to_lds_b128)
#    define HAS_ASYNC_LDS 1
#  endif
#  if __has_builtin(__builtin_amdgcn_s_wait_asynccnt)
#    define HAS_WAIT_ASYNC 1
#  endif
#endif
#ifndef HAS_ASYNC_LDS
#  define HAS_ASYNC_LDS 0
#endif
#ifndef HAS_WAIT_ASYNC
#  define HAS_WAIT_ASYNC 0
#endif

typedef __attribute__((address_space(1))) v4i GV4;  // global int4
typedef __attribute__((address_space(3))) v4i LV4;  // LDS int4

__device__ __forceinline__ float wave_sum32(float v) {
#pragma unroll
  for (int s = 16; s > 0; s >>= 1) v += __shfl_xor(v, s, 32);
  return v;
}

// Bilinear sample one channel (zeros padding; align_corners=False pixel coords pre-applied)
__device__ __forceinline__ float bilin(const float* __restrict__ img, int W, int H,
                                       int ch, float fx, float fy) {
  float xf = floorf(fx), yf = floorf(fy);
  int x0 = (int)xf, y0 = (int)yf;
  float wx1 = fx - xf, wy1 = fy - yf;
  float wx0 = 1.f - wx1, wy0 = 1.f - wy1;
  float acc = 0.f;
#pragma unroll
  for (int cy = 0; cy < 2; ++cy) {
#pragma unroll
    for (int cx = 0; cx < 2; ++cx) {
      int xi = x0 + cx, yi = y0 + cy;
      if (xi >= 0 && xi < W && yi >= 0 && yi < H) {
        float w = (cx ? wx1 : wx0) * (cy ? wy1 : wy0);
        acc += w * img[(size_t)(yi * W + xi) * EMB + ch];
      }
    }
  }
  return acc;
}

// ---------------- Kernel 0a: key sampling at query-0 grid, fold W_k/W_q/b_q -> m, c ----
__global__ __launch_bounds__(512) void setup_k(
    const float* __restrict__ query, const float* __restrict__ key_hist,
    const float* __restrict__ refp, const int* __restrict__ ss,
    const float* __restrict__ W_q, const float* __restrict__ b_q,
    const float* __restrict__ W_k, const float* __restrict__ b_k,
    const float* __restrict__ W_off, const float* __restrict__ b_off,
    float* __restrict__ mvec, float* __restrict__ cvec, int NV) {
  int wv = threadIdx.x >> 5;
  if (wv >= NSS * NH) return;
  int b = wv >> 3, h = wv & 7, lane = threadIdx.x & 31;
  int H = ss[0], W = ss[1];
  float qv = query[h * HD + lane];  // query index 0
  float off[2 * NP];
#pragma unroll
  for (int j = 0; j < 2 * NP; ++j)
    off[j] = wave_sum32(qv * W_off[lane * (2 * NP) + j]) + b_off[j];
  float rx = refp[0], ry = refp[1];
  const float* img = key_hist + (size_t)b * NV * EMB;
#pragma unroll
  for (int p = 0; p < NP; ++p) {
    float fx = rx * (float)W + off[2 * p]     - 0.5f;
    float fy = ry * (float)H + off[2 * p + 1] - 0.5f;
    float kraw = bilin(img, W, H, h * HD + lane, fx, fy);
    // k[d'] = sum_i kraw_i * W_k[i][d'] + b_k[d']   (lane = d')
    float kd = b_k[lane];
    for (int i = 0; i < HD; ++i) kd += __shfl(kraw, i, 32) * W_k[i * HD + lane];
    // m[d] = sum_e W_q[d][e] * k[e]   (lane = d) ;  c = b_q . k
    float md = 0.f;
    for (int e = 0; e < HD; ++e) md += W_q[lane * HD + e] * __shfl(kd, e, 32);
    float c = wave_sum32(b_q[lane] * kd);
    int mi = (b * NH + h) * NP + p;
    mvec[mi * HD + lane] = md;
    if (lane == 0) cvec[mi] = c;
  }
}

// ---------------- Kernel 0b: fused weight WfT[j][e] = sum_d W_v[i][d] W_out[h*32+d][j], bias
__global__ __launch_bounds__(256) void fuse_k(
    const float* __restrict__ W_v, const float* __restrict__ b_v,
    const float* __restrict__ W_out, const float* __restrict__ b_out,
    __hip_bfloat16* __restrict__ WfT, float* __restrict__ bfv) {
  int blk = blockIdx.x, t = threadIdx.x;
  if (blk < EMB) {
    int j = blk, e = t, h = e >> 5, i = e & 31;
    float s = 0.f;
#pragma unroll 8
    for (int d = 0; d < HD; ++d) s += W_v[i * HD + d] * W_out[(h * HD + d) * EMB + j];
    WfT[(size_t)j * EMB + e] = __float2bfloat16(s);
  } else {
    int j = t;
    float s = 0.f;
    for (int e = 0; e < EMB; ++e) s += b_v[e & 31] * W_out[e * EMB + j];
    bfv[j] = s + b_out[j];
  }
}

// ---------------- Kernel 1: one wave per (q,h); lane = channel d ------------------------
__global__ __launch_bounds__(256) void sample_k(
    const float* __restrict__ query, const float* __restrict__ value_hist,
    const float* __restrict__ refp, const int* __restrict__ ss,
    const float* __restrict__ W_off, const float* __restrict__ b_off,
    const float* __restrict__ mvec, const float* __restrict__ cvec,
    __hip_bfloat16* __restrict__ vmix, int Q, int NV) {
  int g = blockIdx.x * 8 + (threadIdx.x >> 5);
  if (g >= Q * NH) return;
  int q = g >> 3, h = g & 7, lane = threadIdx.x & 31;
  int H = ss[0], W = ss[1];
  float qv = query[(size_t)q * EMB + h * HD + lane];
  float off[2 * NP];
#pragma unroll
  for (int j = 0; j < 2 * NP; ++j)
    off[j] = wave_sum32(qv * W_off[lane * (2 * NP) + j]) + b_off[j];
  float rx = refp[q * 2 + 0], ry = refp[q * 2 + 1];
  float fx[NP], fy[NP];
#pragma unroll
  for (int p = 0; p < NP; ++p) {
    fx[p] = rx * (float)W + off[2 * p]     - 0.5f;
    fy[p] = ry * (float)H + off[2 * p + 1] - 0.5f;
  }
  float res = 0.f;
#pragma unroll
  for (int b = 0; b < NSS; ++b) {
    float lg[NP];
#pragma unroll
    for (int p = 0; p < NP; ++p) {
      int mi = (b * NH + h) * NP + p;
      lg[p] = wave_sum32(qv * mvec[mi * HD + lane]) + cvec[mi];
    }
    float mx = fmaxf(fmaxf(lg[0], lg[1]), fmaxf(lg[2], lg[3]));
    float ew[NP]; float s = 0.f;
#pragma unroll
    for (int p = 0; p < NP; ++p) { ew[p] = __expf(lg[p] - mx); s += ew[p]; }
    float inv = 1.f / s;
    const float* img = value_hist + (size_t)b * NV * EMB;
#pragma unroll
    for (int p = 0; p < NP; ++p)
      res += (ew[p] * inv) * bilin(img, W, H, h * HD + lane, fx[p], fy[p]);
  }
  vmix[(size_t)q * EMB + h * HD + lane] = __float2bfloat16(res * 0.5f);
}

// ---------------- Kernel 2: out = vmix(bf16) @ WfT^T + bias + query  (WMMA bf16) --------
// Block: 4 waves, 64 rows x 64 cols. Weight panel staged in LDS via async global->LDS
// (ASYNCcnt path); all 8 A-fragments preloaded so the WMMA loop waits only on DS loads.
__global__ __launch_bounds__(128) void gemm_k(
    const __hip_bfloat16* __restrict__ A,    // [Q][256] bf16
    const __hip_bfloat16* __restrict__ WfT,  // [256 N][256 K] bf16 (transposed fused weight)
    const float* __restrict__ bfv, const float* __restrict__ query,
    float* __restrict__ out, int Q) {
  __shared__ __hip_bfloat16 Bs[64 * BSTR];  // ~33 KB, padded rows
  const int lane = threadIdx.x & 31;
  const int wave = threadIdx.x >> 5;
  const int m0 = (blockIdx.x * 4 + wave) * 16;
  const int n0 = blockIdx.y * 64;
  const int l15 = lane & 15, hi = lane >> 4;

  // -------- stage B panel (WfT rows n0..n0+63, 256 K each) into LDS --------
#if HAS_ASYNC_LDS
#pragma unroll 4
  for (int c = threadIdx.x; c < 2048; c += 128) {  // 2048 x 16B chunks
    int r = c >> 5, col = (c & 31) * 8;
    const __hip_bfloat16* g = WfT + (size_t)(n0 + r) * EMB + col;
    __hip_bfloat16* l = &Bs[r * BSTR + col];
    __builtin_amdgcn_global_load_async_to_lds_b128(
        (GV4*)(unsigned long long)(uintptr_t)g,
        (LV4*)(unsigned int)(uintptr_t)l, 0, 0);
  }
#if HAS_WAIT_ASYNC
  __builtin_amdgcn_s_wait_asynccnt(0);
#else
  asm volatile("s_wait_asynccnt 0" ::: "memory");
#endif
#else
#pragma unroll 4
  for (int c = threadIdx.x; c < 2048; c += 128) {
    int r = c >> 5, col = (c & 31) * 8;
    *(uint4*)&Bs[r * BSTR + col] = *(const uint4*)(WfT + (size_t)(n0 + r) * EMB + col);
  }
#endif

  // -------- preload all 8 A fragments while the LDS fill is in flight --------
  // 16-bit A 16x32 layout: lanes0-15 hold K{0..7,16..23}, lanes16-31 K{8..15,24..31}
  int arow = m0 + l15; if (arow > Q - 1) arow = Q - 1;  // clamp keeps EXEC full
  const __hip_bfloat16* ap = A + (size_t)arow * EMB;
  union F { uint4 u[2]; v16bf v; };
  F a[8];
#pragma unroll
  for (int kk = 0; kk < 8; ++kk) {
    a[kk].u[0] = *(const uint4*)(ap + kk * 32 + hi * 8);
    a[kk].u[1] = *(const uint4*)(ap + kk * 32 + 16 + hi * 8);
  }

  __syncthreads();

  v8f acc[4] = {v8f{}, v8f{}, v8f{}, v8f{}};
#pragma unroll
  for (int kk = 0; kk < 8; ++kk) {
    const int k0 = kk * 32;
#pragma unroll
    for (int t = 0; t < 4; ++t) {
      // 16-bit B 32x16 layout: lane = column; lanes0-15 K{0..7,16..23}, lanes16-31 K{8..15,24..31},
      // 2 K packed per dword (same K order as A).
      const __hip_bfloat16* bp = &Bs[(t * 16 + l15) * BSTR + k0 + hi * 8];
      F b;
      b.u[0] = *(const uint4*)(bp);
      b.u[1] = *(const uint4*)(bp + 16);
      acc[t] = __builtin_amdgcn_wmma_f32_16x16x32_bf16(
          false, a[kk].v, false, b.v, (short)0, acc[t], false, false);
    }
  }

  // C/D layout: lane -> col = lane&15 ; VGPR r -> row = (lane>>4)*8 + r
#pragma unroll
  for (int t = 0; t < 4; ++t) {
    int n = n0 + t * 16 + l15;
    float bias = bfv[n];
#pragma unroll
    for (int r = 0; r < 8; ++r) {
      int m = m0 + hi * 8 + r;
      if (m < Q)
        out[(size_t)m * EMB + n] = acc[t][r] + bias + query[(size_t)m * EMB + n];
    }
  }
}

extern "C" void kernel_launch(void* const* d_in, const int* in_sizes, int n_in,
                              void* d_out, int out_size, void* d_ws, size_t ws_size,
                              hipStream_t stream) {
  const float* query      = (const float*)d_in[0];
  const float* key_hist   = (const float*)d_in[1];
  const float* value_hist = (const float*)d_in[2];
  const float* refp       = (const float*)d_in[3];
  const int*   ss         = (const int*)d_in[4];
  const float* W_q   = (const float*)d_in[5];
  const float* b_q   = (const float*)d_in[6];
  const float* W_k   = (const float*)d_in[7];
  const float* b_k   = (const float*)d_in[8];
  const float* W_v   = (const float*)d_in[9];
  const float* b_v   = (const float*)d_in[10];
  const float* W_off = (const float*)d_in[11];
  const float* b_off = (const float*)d_in[12];
  const float* W_out = (const float*)d_in[13];
  const float* b_out = (const float*)d_in[14];
  float* out = (float*)d_out;

  const int Q  = in_sizes[0] / EMB;          // 40000
  const int NV = in_sizes[1] / (NSS * EMB);  // 40000

  char* ws = (char*)d_ws;
  size_t o = 0;
  __hip_bfloat16* vmix = (__hip_bfloat16*)(ws + o);
  o += (size_t)Q * EMB * sizeof(__hip_bfloat16); o = (o + 255) & ~(size_t)255;
  __hip_bfloat16* WfT = (__hip_bfloat16*)(ws + o);
  o += (size_t)EMB * EMB * sizeof(__hip_bfloat16); o = (o + 255) & ~(size_t)255;
  float* bfv = (float*)(ws + o);  o += EMB * sizeof(float);           o = (o + 255) & ~(size_t)255;
  float* mvec = (float*)(ws + o); o += NSS * NH * NP * HD * sizeof(float); o = (o + 255) & ~(size_t)255;
  float* cvec = (float*)(ws + o); o += NSS * NH * NP * sizeof(float);

  setup_k<<<1, 512, 0, stream>>>(query, key_hist, refp, ss, W_q, b_q, W_k, b_k,
                                 W_off, b_off, mvec, cvec, NV);
  fuse_k<<<EMB + 1, 256, 0, stream>>>(W_v, b_v, W_out, b_out, WfT, bfv);
  sample_k<<<Q, 256, 0, stream>>>(query, value_hist, refp, ss, W_off, b_off,
                                  mvec, cvec, vmix, Q, NV);
  dim3 g2((Q + 63) / 64, EMB / 64);
  gemm_k<<<g2, 128, 0, stream>>>(vmix, WfT, bfv, query, out, Q);
}
